// LocalAggregator_67937792688246
// MI455X (gfx1250) — matrix operand order repaired and checked by reading, not compiled, not run on hardware
//
#include <hip/hip_runtime.h>

typedef __attribute__((ext_vector_type(16))) _Float16 v16h;
typedef __attribute__((ext_vector_type(8)))  _Float16 v8h;
typedef __attribute__((ext_vector_type(8)))  float    v8f;
typedef __attribute__((ext_vector_type(4)))  unsigned int u32x4;
typedef __attribute__((ext_vector_type(8)))  int          i32x8;
typedef __attribute__((ext_vector_type(4)))  int          i32x4;

#define B_    128
#define N_    100
#define NP    112      // N padded to 7*16
#define D_    128
#define NT    7        // number of 16-row tiles covering N
#define K12   12       // 3 (A) + 9 (Bm) score channels
#define ALPHA 0.2f
#define MASKV -9.0e15f

#define RS    136      // padded LDS row stride in halves (272B = 68 dwords -> conflict-free)
#define AIKS  (16 * RS)

// dynamic-LDS layout (bytes); no static __shared__ anywhere -> dynamic base = 0
#define OFF_HT    0        // _Float16 HT[128][RS]   (h transposed, d-major)   34816
#define OFF_HA    34816    // _Float16 HA[128][RS]   (h row-major, j-major)    34816
#define OFF_AIK   69632    // _Float16 AIK[12][16][RS]                         52224
#define OFF_STAGE 121856   // float STAGE[100][128]  TDM landing zone          51200
// ---- the following alias STAGE (live only after GEMM1 consumed STAGE) ----
#define OFF_SE    121856   // float SE[16][112]                                 7168
#define OFF_SB    129024   // float SB[16][112]                                 7168
#define OFF_P     136192   // _Float16 P[16][RS]                                4352
#define OFF_PMAX  140544   // float PMAX[2][16][8]                              1024
#define OFF_RSUM  141568   // float RSUM[2*16]                                   128
#define OFF_RMAX  141696   // float RMAX[2*16]                                   128
#define LDS_TOTAL 173056   // OFF_STAGE + 51200

__global__ __launch_bounds__(224)
__attribute__((amdgpu_waves_per_eu(1)))          // LDS-bound occupancy: spend VGPRs freely
void gat_fused_wmma(const float* __restrict__ hidden,
                    const int*   __restrict__ adj,
                    const int*   __restrict__ beh,
                    const float* __restrict__ Aw,
                    const float* __restrict__ Bw,
                    float*       __restrict__ out)
{
    extern __shared__ char smem[];
    _Float16* HT   = (_Float16*)(smem + OFF_HT);
    _Float16* HA   = (_Float16*)(smem + OFF_HA);
    _Float16* AIK  = (_Float16*)(smem + OFF_AIK);
    const float* SG = (const float*)(smem + OFF_STAGE);
    float*    SE   = (float*)(smem + OFF_SE);
    float*    SB   = (float*)(smem + OFF_SB);
    _Float16* P    = (_Float16*)(smem + OFF_P);
    float*    PMAX = (float*)(smem + OFF_PMAX);
    float*    RSUM = (float*)(smem + OFF_RSUM);
    float*    RMAX = (float*)(smem + OFF_RMAX);

    const int tid  = threadIdx.x;
    const int lane = tid & 31;
    const int wave = tid >> 5;            // 0..6
    const int b    = blockIdx.x / NT;
    const int it   = blockIdx.x % NT;
    const int i0   = it * 16;

    const float* Hb = hidden + b * N_ * D_;

    // ---- TDM: DMA hidden[b] (100x128 f32 tile) into LDS STAGE ----
    if (wave == 0) {
        const unsigned long long ga = (unsigned long long)(const void*)Hb;
        u32x4 g0;
        g0[0] = 1u;                                        // count=1, user descriptor
        g0[1] = (unsigned)OFF_STAGE;                       // lds_addr
        g0[2] = (unsigned)(ga & 0xFFFFFFFFu);              // global_addr[31:0]
        g0[3] = ((unsigned)(ga >> 32) & 0x01FFFFFFu)       // global_addr[56:32]
              | (2u << 30);                                // type = 2 (image)
        i32x8 g1;
        g1[0] = (2 << 16);                                 // data_size = 4 bytes
        g1[1] = (int)(128u << 16);                         // tensor_dim0 = 128
        g1[2] = (int)(100u << 16);                         // tensor_dim1 = 100
        g1[3] = (int)(128u << 16);                         // tile_dim0 = 128
        g1[4] = 100;                                       // tile_dim1 = 100, tile_dim2 = 0
        g1[5] = 128;                                       // tensor_dim0_stride = 128
        g1[6] = 0;
        g1[7] = 0;
        i32x4 gz4 = {0, 0, 0, 0};                          // 2D: groups 2/3 unused
        i32x8 gz8 = {0, 0, 0, 0, 0, 0, 0, 0};
        __builtin_amdgcn_tensor_load_to_lds(g0, g1, gz4, gz4, gz8, 0);
        __builtin_amdgcn_s_wait_tensorcnt(0);
    }
    __syncthreads();

    // ---- convert STAGE(f32) -> HA / HT (f16, zero-padded, stride RS) ----
    for (int idx = tid; idx < 128 * 128; idx += 224) {
        const int j = idx >> 7, d = idx & 127;
        const float v = (j < N_) ? SG[j * 128 + d] : 0.0f;
        const _Float16 hv = (_Float16)v;
        HA[j * RS + d] = hv;       // B operand of GEMM1 (K=d along row j)
        HT[d * RS + j] = hv;       // B operand of GEMM2 (K=j along row d)
    }
    // ---- AIK[k][m][d] = h[i0+m][d] * C[d][k]  (from LDS stage) ----
    for (int idx = tid; idx < K12 * 16 * D_; idx += 224) {
        const int k = idx >> 11;
        const int r = idx & 2047;
        const int m = r >> 7, d = r & 127;
        const int i = i0 + m;
        const float hv = (i < N_) ? SG[i * 128 + d] : 0.0f;
        const float c  = (k < 3) ? Aw[d * 3 + k] : Bw[d * 9 + (k - 3)];
        AIK[k * AIKS + m * RS + d] = (_Float16)(hv * c);
    }
    __syncthreads();

    // =====================  GEMM1: 12 score tiles per wave, 2 passes of 6  =====================
    {
        const int jt = wave;
        const int m  = lane & 15;          // A-operand row (M)
        const int hh = lane >> 4;          // A-operand K-half selector
        const int n  = lane & 15;          // D-matrix column (N)
        const int mb = (lane >> 4) * 8;    // D-matrix row base
        const int jcol = jt * 16 + n;

        // fetch adjacency codes up front: latency hidden behind both GEMM passes
        int ca8[8], cb8[8];
#pragma unroll
        for (int r = 0; r < 8; ++r) {
            const int i = i0 + mb + r;
            ca8[r] = 0; cb8[r] = 0;
            if (jcol < N_ && i < N_) {
                ca8[r] = __builtin_nontemporal_load(&adj[(b * N_ + i) * N_ + jcol]);
                cb8[r] = __builtin_nontemporal_load(&beh[(b * N_ + i) * N_ + jcol]);
            }
        }

        auto loadA = [&](int ks, int k) -> v16h {
            const _Float16* ap = &AIK[k * AIKS + m * RS + ks * 32 + hh * 8];
            const v8h lo = *(const v8h*)ap;          // 16B-aligned
            const v8h hi = *(const v8h*)(ap + 16);
            return __builtin_shufflevector(lo, hi,
                0,1,2,3,4,5,6,7,8,9,10,11,12,13,14,15);
        };
        auto loadB = [&](int ks) -> v16h {
            const _Float16* bp = &HA[(jt * 16 + n) * RS + ks * 32 + hh * 8];
            const v8h lo = *(const v8h*)bp;
            const v8h hi = *(const v8h*)(bp + 16);
            return __builtin_shufflevector(lo, hi,
                0,1,2,3,4,5,6,7,8,9,10,11,12,13,14,15);
        };

        float xe[8], xb[8];
#pragma unroll
        for (int r = 0; r < 8; ++r) { xe[r] = 0.f; xb[r] = 0.f; }

#pragma unroll
        for (int p = 0; p < 2; ++p) {                  // channels p*6 .. p*6+5
            v8f acc[6] = {};
            v16h areg[6];
#pragma unroll
            for (int k6 = 0; k6 < 6; ++k6) areg[k6] = loadA(0, p * 6 + k6);
            v16h b_cur = loadB(0);
#pragma unroll
            for (int ks = 0; ks < 4; ++ks) {
                // issue ALL loads for step ks+1 first; barrier pins them above the WMMAs
                v16h b_nxt = b_cur;
                v16h anew[6];
                if (ks < 3) {
                    b_nxt = loadB(ks + 1);
#pragma unroll
                    for (int k6 = 0; k6 < 6; ++k6) anew[k6] = loadA(ks + 1, p * 6 + k6);
                }
                __builtin_amdgcn_sched_barrier(0);
#pragma unroll
                for (int k6 = 0; k6 < 6; ++k6) {
                    acc[k6] = __builtin_amdgcn_wmma_f32_16x16x32_f16(
                        false, areg[k6], false, b_cur, (short)0, acc[k6], false, false);
                }
                __builtin_amdgcn_sched_barrier(0);
                if (ks < 3) {
#pragma unroll
                    for (int k6 = 0; k6 < 6; ++k6) areg[k6] = anew[k6];
                    b_cur = b_nxt;
                }
            }
            // channel->code mapping: pass0 = e codes 1..3 (k6 0..2), b codes 1..3 (k6 3..5)
            //                        pass1 = b codes 4..9 (k6 0..5)
#pragma unroll
            for (int r = 0; r < 8; ++r) {
                if (p == 0) {
#pragma unroll
                    for (int k6 = 0; k6 < 3; ++k6)
                        if (ca8[r] == k6 + 1) xe[r] = acc[k6][r];
#pragma unroll
                    for (int k6 = 3; k6 < 6; ++k6)
                        if (cb8[r] == k6 - 2) xb[r] = acc[k6][r];
                } else {
#pragma unroll
                    for (int k6 = 0; k6 < 6; ++k6)
                        if (cb8[r] == k6 + 4) xb[r] = acc[k6][r];
                }
            }
        }

        // ---- LeakyReLU + mask -> SE / SB (aliases STAGE: safe, synced) ----
#pragma unroll
        for (int r = 0; r < 8; ++r) {
            const int M = mb + r;
            const int i = i0 + M;
            float se = MASKV, sb = MASKV;
            if (jcol >= N_) {
                se = -__builtin_inff();
                sb = -__builtin_inff();
            } else if (i < N_) {
                if (ca8[r] >= 1 && ca8[r] <= 3) se = (xe[r] >= 0.f) ? xe[r] : ALPHA * xe[r];
                if (cb8[r] >= 1 && cb8[r] <= 9) sb = (xb[r] >= 0.f) ? xb[r] : ALPHA * xb[r];
            }
            SE[M * NP + jcol] = se;
            SB[M * NP + jcol] = sb;
        }
    }
    __syncthreads();

    // ===========  row softmax, parallel over 224 lanes (2 maps x 16 rows x 7 segs) ===========
    {
        const int map = (tid >= 112) ? 1 : 0;
        const int tt  = tid - 112 * map;
        const int row = tt / 7;
        const int seg = tt % 7;
        float* S  = map ? SB : SE;
        float* PM = PMAX + map * 128 + row * 8;
        const int base = row * NP + seg * 16;

        float mx = -3.4e38f;
#pragma unroll
        for (int c = 0; c < 16; ++c) mx = fmaxf(mx, S[base + c]);
        PM[seg] = mx;
        __syncthreads();
        if (seg == 0) {
            float m2 = PM[0];
#pragma unroll
            for (int s = 1; s < 7; ++s) m2 = fmaxf(m2, PM[s]);
            RMAX[map * 16 + row] = m2;
        }
        __syncthreads();
        const float rm = RMAX[map * 16 + row];
        float sum = 0.f;
#pragma unroll
        for (int c = 0; c < 16; ++c) {
            const float e = __expf(S[base + c] - rm);
            S[base + c] = e;                 // store exp in place
            sum += e;
        }
        PM[seg] = sum;
        __syncthreads();
        if (seg == 0) {
            float s2 = 0.f;
#pragma unroll
            for (int s = 0; s < 7; ++s) s2 += PM[s];
            RSUM[map * 16 + row] = 1.0f / s2;
        }
        __syncthreads();
    }

    // ---- P[i][j] = 0.5*(alpha + alpha_beh), f16, zero-padded to K=128 ----
    for (int idx = tid; idx < 16 * 128; idx += 224) {
        const int i = idx >> 7, j = idx & 127;
        _Float16 pv = (_Float16)0.0f;
        if (j < NP) {
            pv = (_Float16)(0.5f * (SE[i * NP + j] * RSUM[i] +
                                    SB[i * NP + j] * RSUM[16 + i]));
        }
        P[i * RS + j] = pv;
    }
    __syncthreads();

    // =====================  GEMM2: out = P x H  =====================
    {
        const int m  = lane & 15;
        const int hh = lane >> 4;
        const int n  = lane & 15;
        const int mb = (lane >> 4) * 8;

        auto loadB2 = [&](int ks, int dt) -> v16h {
            const _Float16* bp = &HT[(dt * 16 + n) * RS + ks * 32 + hh * 8];
            const v8h lo = *(const v8h*)bp;
            const v8h hi = *(const v8h*)(bp + 16);
            return __builtin_shufflevector(lo, hi,
                0,1,2,3,4,5,6,7,8,9,10,11,12,13,14,15);
        };

        v16h pa[4];
#pragma unroll
        for (int ks = 0; ks < 4; ++ks) {
            const _Float16* ap = &P[m * RS + ks * 32 + hh * 8];
            const v8h lo = *(const v8h*)ap;
            const v8h hi = *(const v8h*)(ap + 16);
            pa[ks] = __builtin_shufflevector(lo, hi,
                0,1,2,3,4,5,6,7,8,9,10,11,12,13,14,15);
        }
        for (int dt = wave; dt < 8; dt += 7) {
            v8f o = {};
            v16h b_cur = loadB2(0, dt);
#pragma unroll
            for (int ks = 0; ks < 4; ++ks) {
                const v16h b_next = (ks < 3) ? loadB2(ks + 1, dt) : b_cur;
                o = __builtin_amdgcn_wmma_f32_16x16x32_f16(
                    false, pa[ks], false, b_cur, (short)0, o, false, false);
                b_cur = b_next;
            }
#pragma unroll
            for (int r = 0; r < 8; ++r) {
                const int i = i0 + mb + r;
                if (i < N_) {
                    __builtin_nontemporal_store(o[r], &out[(b * N_ + i) * D_ + dt * 16 + n]);
                }
            }
        }
    }
}

extern "C" void kernel_launch(void* const* d_in, const int* in_sizes, int n_in,
                              void* d_out, int out_size, void* d_ws, size_t ws_size,
                              hipStream_t stream) {
    (void)in_sizes; (void)n_in; (void)d_ws; (void)ws_size; (void)out_size;
    const float* hidden = (const float*)d_in[0];
    const int*   adj    = (const int*)  d_in[1];
    const int*   beh    = (const int*)  d_in[2];
    const float* Aw     = (const float*)d_in[3];
    const float* Bw     = (const float*)d_in[4];
    float*       out    = (float*)d_out;

    (void)hipFuncSetAttribute((const void*)gat_fused_wmma,
                              hipFuncAttributeMaxDynamicSharedMemorySize, LDS_TOTAL);

    dim3 grid(B_ * NT);   // 128 batches x 7 i-tiles = 896 workgroups
    dim3 block(224);      // 7 wave32 waves
    gat_fused_wmma<<<grid, block, LDS_TOTAL, stream>>>(hidden, adj, beh, Aw, Bw, out);
}
